// MyLSTM_69114613729315
// MI455X (gfx1250) — matrix operand order repaired and checked by reading, not compiled, not run on hardware
//
#include <hip/hip_runtime.h>

typedef __attribute__((ext_vector_type(16))) __bf16 v16bf;
typedef __attribute__((ext_vector_type(8)))  __bf16 v8bf;
typedef __attribute__((ext_vector_type(8)))  float  v8f;

constexpr int SEQ = 2048, BATCH = 48, INDIM = 128, H = 256, GATES = 4 * H;
constexpr int CH = 256;  // timesteps per chunk (8 chunks)
constexpr int LT_STRIDE = 3 * 8 * 8 * 256;  // Xg floats per timestep

__device__ __forceinline__ float sigf(float x) { return 1.0f / (1.0f + __expf(-x)); }
// Branch-free tanh: 1 - 2/(exp(2x)+1). exp->inf => 1; exp->0 => -1. No OCML call.
__device__ __forceinline__ float tanh_fast(float x) {
  return 1.0f - 2.0f / (__expf(2.0f * x) + 1.0f);
}
// LICM fence: keeps the loop-invariant Whh fragment loads inside the timestep
// loop (hoisting 32 fragments spills them to scratch).
__device__ __forceinline__ void mem_fence_opt() { asm volatile("" ::: "memory"); }

__global__ void conv_bf16_kernel(const float* __restrict__ s, __bf16* __restrict__ d, int n) {
  for (int i = blockIdx.x * blockDim.x + threadIdx.x; i < n; i += gridDim.x * blockDim.x)
    d[i] = (__bf16)s[i];
}

__global__ void bias_sum_kernel(const float* __restrict__ a, const float* __restrict__ b,
                                float* __restrict__ o, int n) {
  int i = blockIdx.x * blockDim.x + threadIdx.x;
  if (i < n) o[i] = a[i] + b[i];
}

// ---------------------------------------------------------------------------
// Full-GPU input projection for one chunk: Xg[t'] = x_{t0+t'} @ Wih^T + bias,
// stored in WMMA C-fragment order [t'][mt][cg8][g][j][lane][8] so the serial
// recurrence kernel can load each accumulator tile as 32 contiguous bytes/lane.
// One block per timestep; 24 waves = (mt 0..2) x (cg8 0..7), wave owns 8 tiles.
// ---------------------------------------------------------------------------
template <int KIN>
__global__ __launch_bounds__(768, 1) void gates_gemm_kernel(
    const __bf16* __restrict__ Xsrc,  // [CH, BATCH, KIN] (chunk base)
    const __bf16* __restrict__ Wih,   // [GATES, KIN]
    const float* __restrict__ bias,   // [GATES]
    float* __restrict__ Xg) {         // fragment layout
  const int tid  = threadIdx.x;
  const int wave = tid >> 5;
  const int lane = tid & 31;
  const int mt   = wave % 3;
  const int cg   = wave / 3;
  const int nb   = cg * 32;
  const int n    = lane & 15;
  const int hi   = lane >> 4;
  const int kb   = hi * 16;
  const int bkb  = hi * 16;
  const int arow = mt * 16 + n;

  const __bf16* xr = Xsrc + (size_t)(blockIdx.x * BATCH + arow) * KIN;
  const __bf16* wl = Wih + (size_t)(nb + n) * KIN + bkb;

  v8f C[4][2];
#pragma unroll
  for (int g = 0; g < 4; ++g)
#pragma unroll
    for (int j = 0; j < 2; ++j) {
      float bvv = bias[g * H + nb + j * 16 + n];
#pragma unroll
      for (int r = 0; r < 8; ++r) C[g][j][r] = bvv;
    }

#pragma unroll
  for (int ks = 0; ks < KIN / 32; ++ks) {
    v16bf a;
    *((v8bf*)&a)     = *(const v8bf*)(xr + ks * 32 + kb);
    *((v8bf*)&a + 1) = *(const v8bf*)(xr + ks * 32 + kb + 8);
#pragma unroll
    for (int g = 0; g < 4; ++g)
#pragma unroll
      for (int j = 0; j < 2; ++j) {
        v16bf b = *(const v16bf*)(wl + ((size_t)g * H + j * 16) * KIN + ks * 32);
        C[g][j] = __builtin_amdgcn_wmma_f32_16x16x32_bf16(
            false, a, false, b, (short)0, C[g][j], false, false);
      }
  }

  float* xo = Xg + ((((size_t)blockIdx.x * 3 + mt) * 8 + cg) * 8) * 256 + lane * 8;
#pragma unroll
  for (int g = 0; g < 4; ++g)
#pragma unroll
    for (int j = 0; j < 2; ++j) *(v8f*)(xo + (g * 2 + j) * 256) = C[g][j];
}

// ---------------------------------------------------------------------------
// Serial recurrence for one chunk: ONE workgroup, 16 waves (512 threads).
// wave = cg16 (16-column group of H). Each wave owns all 4 gate tiles x all 3
// M-tiles for its 16 h-columns: every Whh fragment is fetched by exactly one
// wave (no duplicate L2 traffic) and feeds 3 WMMAs. Activations + cell state
// stay in registers; h lives in LDS (bf16); both persist across chunks via
// tiny global buffers.
// ---------------------------------------------------------------------------
template <bool WRITE_Y, bool WRITE_LAST>
__global__ __launch_bounds__(512, 1) void lstm_rec_kernel(
    const float* __restrict__ Xg,      // chunk gate fragments (incl. bias)
    const __bf16* __restrict__ Whh,    // [GATES, H]
    __bf16* __restrict__ yout,         // chunk base [CH, BATCH, H] (layer0)
    float* __restrict__ lastrow,       // chunk base [CH, H] (layer1)
    float* __restrict__ hfinal,        // [BATCH, H]
    __bf16* __restrict__ hstate,       // [BATCH*H] carried h
    float* __restrict__ cstate,        // [BATCH*H] carried c
    int t0, int steps, int Stot) {
  __shared__ __bf16 hb[BATCH * H];  // 24 KB

  const int tid  = threadIdx.x;
  const int cg16 = tid >> 5;         // wave 0..15: h columns [cg16*16, +16)
  const int lane = tid & 31;
  const int n    = lane & 15;
  const int hi   = lane >> 4;
  const int kb   = hi * 16;          // A-fragment k base (contiguous k = 16*hi + s)
  const int bkb  = hi * 16;          // B-fragment k base
  const int col  = cg16 * 16 + n;    // C column (same for all tiles)

  // Xg tile offsets for this wave: tile (g, mt) at cg8 = cg16>>1, j = cg16&1.
  int xg_off[4][3];
#pragma unroll
  for (int g = 0; g < 4; ++g)
#pragma unroll
    for (int mt = 0; mt < 3; ++mt)
      xg_off[g][mt] = (((mt * 8 + (cg16 >> 1)) * 8) + g * 2 + (cg16 & 1)) * 256;

  v8f cst[3];
  if (t0 == 0) {
    for (int i = tid; i < BATCH * H; i += 512) hb[i] = (__bf16)0.0f;
#pragma unroll
    for (int mt = 0; mt < 3; ++mt)
#pragma unroll
      for (int r = 0; r < 8; ++r) cst[mt][r] = 0.0f;
  } else {
    for (int i = tid; i < BATCH * H; i += 512) hb[i] = hstate[i];
#pragma unroll
    for (int mt = 0; mt < 3; ++mt)
#pragma unroll
      for (int r = 0; r < 8; ++r) cst[mt][r] = cstate[(mt * 16 + hi * 8 + r) * H + col];
  }

  // Per-lane Whh base: column (g*H + col), contiguous K per lane.
  const __bf16* whh_l = Whh + (size_t)col * H + bkb;  // + g*H*H + ks*32
  const float* xg_w = Xg + lane * 8;                  // + xg_off + lt*LT_STRIDE

  __syncthreads();

  for (int lt = 0; lt < steps; ++lt) {
    mem_fence_opt();  // no LICM of Whh fragments -> no scratch spills

    // ---- accumulators preloaded with x@Wih^T + bias (fragment layout) ----
    v8f C[4][3];
#pragma unroll
    for (int g = 0; g < 4; ++g)
#pragma unroll
      for (int mt = 0; mt < 3; ++mt) C[g][mt] = *(const v8f*)(xg_w + xg_off[g][mt]);

    // prefetch next timestep's gate fragments into the cache hierarchy
    if (lt + 1 < steps) {
#pragma unroll
      for (int g = 0; g < 4; ++g)
#pragma unroll
        for (int mt = 0; mt < 3; ++mt)
          __builtin_prefetch(xg_w + LT_STRIDE + xg_off[g][mt], 0, 1);
    }

    // ---- gates += h_{t-1} @ Whh^T (K = H), A from LDS, B from L2 ----
#pragma unroll
    for (int ks = 0; ks < H / 32; ++ks) {
      v16bf a[3];
#pragma unroll
      for (int mt = 0; mt < 3; ++mt) {
        *((v8bf*)&a[mt])     = *(const v8bf*)(&hb[(mt * 16 + n) * H + ks * 32 + kb]);
        *((v8bf*)&a[mt] + 1) = *(const v8bf*)(&hb[(mt * 16 + n) * H + ks * 32 + kb + 8]);
      }
#pragma unroll
      for (int g = 0; g < 4; ++g) {
        v16bf b = *(const v16bf*)(whh_l + (size_t)g * H * H + ks * 32);
#pragma unroll
        for (int mt = 0; mt < 3; ++mt)
          C[g][mt] = __builtin_amdgcn_wmma_f32_16x16x32_bf16(
              false, a[mt], false, b, (short)0, C[g][mt], false, false);
      }
    }

    __syncthreads();  // all waves done reading hb(t-1)

    // ---- activations + cell update, all in registers ----
#pragma unroll
    for (int mt = 0; mt < 3; ++mt) {
#pragma unroll
      for (int r = 0; r < 8; ++r) {
        float iv = sigf(C[0][mt][r]);
        float fv = sigf(C[1][mt][r]);
        float gv = tanh_fast(C[2][mt][r]);
        float ov = sigf(C[3][mt][r]);
        float cv = fv * cst[mt][r] + iv * gv;
        cst[mt][r] = cv;
        float hv = ov * tanh_fast(cv);
        const int row = mt * 16 + hi * 8 + r;
        __bf16 hx = (__bf16)hv;
        hb[row * H + col] = hx;
        if (WRITE_Y)    yout[(size_t)row * H + col] = hx;
        if (WRITE_LAST) { if (row == BATCH - 1) lastrow[col] = hv; }
        if (t0 + lt == Stot - 1) hfinal[row * H + col] = hv;
      }
    }

    __syncthreads();  // hb(t) visible before next step's A loads

    xg_w += LT_STRIDE;
    if (WRITE_Y)    yout += (size_t)BATCH * H;
    if (WRITE_LAST) lastrow += H;
  }

  // ---- persist h / c state for the next chunk ----
  for (int i = tid; i < BATCH * H; i += 512) hstate[i] = hb[i];
#pragma unroll
  for (int mt = 0; mt < 3; ++mt)
#pragma unroll
    for (int r = 0; r < 8; ++r) cstate[(mt * 16 + hi * 8 + r) * H + col] = cst[mt][r];
}

__global__ void fc_kernel(const float* __restrict__ ylast, const float* __restrict__ w,
                          const float* __restrict__ bb, float* __restrict__ out, int S) {
  int gw   = (blockIdx.x * blockDim.x + threadIdx.x) >> 5;
  int lane = threadIdx.x & 31;
  if (gw >= S) return;
  const float* row = ylast + (size_t)gw * H;
  float s = 0.f;
  for (int k = lane; k < H; k += 32) s += row[k] * w[k];
#pragma unroll
  for (int off = 16; off > 0; off >>= 1) s += __shfl_xor(s, off, 32);
  if (lane == 0) out[gw] = s + bb[0];
}

extern "C" void kernel_launch(void* const* d_in, const int* in_sizes, int n_in,
                              void* d_out, int out_size, void* d_ws, size_t ws_size,
                              hipStream_t stream) {
  (void)in_sizes; (void)n_in; (void)out_size; (void)ws_size;
  const float* x    = (const float*)d_in[0];
  const float* Wih0 = (const float*)d_in[1];
  const float* Whh0 = (const float*)d_in[2];
  const float* bih0 = (const float*)d_in[3];
  const float* bhh0 = (const float*)d_in[4];
  const float* Wih1 = (const float*)d_in[5];
  const float* Whh1 = (const float*)d_in[6];
  const float* bih1 = (const float*)d_in[7];
  const float* bhh1 = (const float*)d_in[8];
  const float* fcw  = (const float*)d_in[9];
  const float* fcb  = (const float*)d_in[10];
  float* out = (float*)d_out;

  char* ws = (char*)d_ws;
  size_t off = 0;
  auto take = [&](size_t bytes) -> char* {
    char* p = ws + off;
    off = (off + bytes + 255) & ~(size_t)255;
    return p;
  };
  __bf16* xbf   = (__bf16*)take((size_t)SEQ * BATCH * INDIM * 2);  // 25 MB
  __bf16* wih0b = (__bf16*)take((size_t)GATES * INDIM * 2);
  __bf16* whh0b = (__bf16*)take((size_t)GATES * H * 2);
  __bf16* wih1b = (__bf16*)take((size_t)GATES * H * 2);
  __bf16* whh1b = (__bf16*)take((size_t)GATES * H * 2);
  float*  bias0 = (float*)take((size_t)GATES * 4);
  float*  bias1 = (float*)take((size_t)GATES * 4);
  __bf16* y0bf  = (__bf16*)take((size_t)SEQ * BATCH * H * 2);      // 50 MB
  float*  ylast = (float*)take((size_t)SEQ * H * 4);               // 2 MB
  float*  Xg    = (float*)take((size_t)CH * BATCH * GATES * 4);    // 50 MB
  __bf16* hst   = (__bf16*)take((size_t)BATCH * H * 2);
  float*  cstt  = (float*)take((size_t)BATCH * H * 4);

  auto cv = [&](const float* s, __bf16* d, int n) {
    int blocks = (n + 2047) / 2048;
    if (blocks > 4096) blocks = 4096;
    conv_bf16_kernel<<<blocks, 256, 0, stream>>>(s, d, n);
  };
  cv(x,    xbf,   SEQ * BATCH * INDIM);
  cv(Wih0, wih0b, GATES * INDIM);
  cv(Whh0, whh0b, GATES * H);
  cv(Wih1, wih1b, GATES * H);
  cv(Whh1, whh1b, GATES * H);
  bias_sum_kernel<<<4, 256, 0, stream>>>(bih0, bhh0, bias0, GATES);
  bias_sum_kernel<<<4, 256, 0, stream>>>(bih1, bhh1, bias1, GATES);

  float* h0n = out + SEQ;             // hn[0] : [48,256]
  float* h1n = out + SEQ + BATCH * H; // hn[1] : [48,256]

  for (int c = 0; c < SEQ / CH; ++c) {
    gates_gemm_kernel<INDIM><<<CH, 768, 0, stream>>>(
        xbf + (size_t)c * CH * BATCH * INDIM, wih0b, bias0, Xg);
    lstm_rec_kernel<true, false><<<1, 512, 0, stream>>>(
        Xg, whh0b, y0bf + (size_t)c * CH * BATCH * H, nullptr, h0n,
        hst, cstt, c * CH, CH, SEQ);
  }
  for (int c = 0; c < SEQ / CH; ++c) {
    gates_gemm_kernel<H><<<CH, 768, 0, stream>>>(
        y0bf + (size_t)c * CH * BATCH * H, wih1b, bias1, Xg);
    lstm_rec_kernel<false, true><<<1, 512, 0, stream>>>(
        Xg, whh1b, nullptr, ylast + (size_t)c * CH * H, h1n,
        hst, cstt, c * CH, CH, SEQ);
  }

  fc_kernel<<<(SEQ * 32 + 255) / 256, 256, 0, stream>>>(ylast, fcw, fcb, out, SEQ);
}
